// SocialLSTM_36086315221294
// MI455X (gfx1250) — matrix operand order repaired and checked by reading, compiled, not run on hardware
//
#include <hip/hip_runtime.h>
#include <hip/hip_bf16.h>

// ---------------------------------------------------------------------------
// Social LSTM on MI455X (gfx1250): wave32 WMMA f16 pipeline.
// T=20, N=256, G2=64, H=128, E=64.
// ten_emb refactored as  grid(256x16384) @ M(16384x64), M = h @ W_t_g^T.
// GEMM2 streams grids (f32, 335 MB total) straight from HBM, converting to
// f16 in-register (0/1 values: exact). GEMM inner loops are software-
// pipelined AND pinned with sched_group_barrier (VMEM-read group for k-step
// n+1, WMMA group for k-step n) so the backend cannot re-serialize them.
// ---------------------------------------------------------------------------

typedef __attribute__((ext_vector_type(16))) _Float16 v16h;
typedef __attribute__((ext_vector_type(8)))  _Float16 v8h;
typedef __attribute__((ext_vector_type(8)))  float    v8f;

#define TT 20
#define NN 256
#define HH 128
#define EE 64
#define G2 64
#define KSOC (NN * G2)         // 16384
#define NCHUNK 64              // K-split chunks for GEMM2 (1024 waves)
#define KCHUNK (KSOC / NCHUNK) // 256 -> 8 k-steps per wave

#define WMMA_F16(a, b, c) \
  __builtin_amdgcn_wmma_f32_16x16x32_f16(false, (a), false, (b), (short)0, (c), false, false)

#if defined(__has_builtin)
#if __has_builtin(__builtin_amdgcn_sched_group_barrier)
#define SCHED_GROUP(mask, size) __builtin_amdgcn_sched_group_barrier((mask), (size), 0)
#endif
#endif
#ifndef SCHED_GROUP
#define SCHED_GROUP(mask, size)
#endif
#define SG_VMEM_RD 0x020
#define SG_WMMA    0x008

// ---- WMMA fragment loaders (wave32 layouts, CDNA5 ISA 7.12.2) ----
// A 16x32 f16: lanes 0-15 -> M=lane, K = {k..k+7, k+16..k+23};
//              lanes 16-31 -> M=lane-16, K = {k+8..k+15, k+24..k+31}.
__device__ __forceinline__ v16h ld_a(const _Float16* __restrict__ A, int lda,
                                     int row, int khalf, int k) {
  const _Float16* p = A + (size_t)row * lda + k + khalf * 8;
  v8h lo = *(const v8h*)(p);
  v8h hi = *(const v8h*)(p + 16);
  return __builtin_shufflevector(lo, hi, 0,1,2,3,4,5,6,7,8,9,10,11,12,13,14,15);
}
// Same fragment, but source data is f32: 4x b128 loads + in-register cvt.
__device__ __forceinline__ v16h ld_a_f32(const float* __restrict__ p) {
  float4 f0 = ((const float4*)p)[0];
  float4 f1 = ((const float4*)p)[1];
  float4 f2 = ((const float4*)(p + 16))[0];
  float4 f3 = ((const float4*)(p + 16))[1];
  v16h a;
  a[0]  = (_Float16)f0.x; a[1]  = (_Float16)f0.y; a[2]  = (_Float16)f0.z; a[3]  = (_Float16)f0.w;
  a[4]  = (_Float16)f1.x; a[5]  = (_Float16)f1.y; a[6]  = (_Float16)f1.z; a[7]  = (_Float16)f1.w;
  a[8]  = (_Float16)f2.x; a[9]  = (_Float16)f2.y; a[10] = (_Float16)f2.z; a[11] = (_Float16)f2.w;
  a[12] = (_Float16)f3.x; a[13] = (_Float16)f3.y; a[14] = (_Float16)f3.z; a[15] = (_Float16)f3.w;
  return a;
}
// B 32x16 f16 stored transposed (BT: N rows x K cols): lanes 0-15 -> N=lane,
// K = {k..k+7, k+16..k+23}; lanes 16-31 -> N=lane-16, K = {k+8..k+15, k+24..k+31}.
__device__ __forceinline__ v16h ld_b(const _Float16* __restrict__ BT, int ldb,
                                     int col, int khalf, int k) {
  const _Float16* p = BT + (size_t)col * ldb + k + khalf * 8;
  v8h lo = *(const v8h*)(p);
  v8h hi = *(const v8h*)(p + 16);
  return __builtin_shufflevector(lo, hi, 0,1,2,3,4,5,6,7,8,9,10,11,12,13,14,15);
}

__device__ __forceinline__ float sigmf(float x) { return 1.0f / (1.0f + __expf(-x)); }

// ---- one-time weight prep: f32 -> f16, concat, bias fold --------------------
__global__ void k_prep(const float* __restrict__ W_t,
                       const float* __restrict__ W_ih, const float* __restrict__ W_hh,
                       const float* __restrict__ b_ih, const float* __restrict__ b_hh,
                       _Float16* __restrict__ Wt_h, _Float16* __restrict__ Wcat_h,
                       float* __restrict__ bias_g) {
  int tid = blockIdx.x * blockDim.x + threadIdx.x;
  if (tid < EE * G2 * HH) Wt_h[tid] = (_Float16)W_t[tid];               // 64 x 8192
  if (tid < 512 * 256) {                                                 // [W_ih | W_hh]
    int j = tid >> 8, kk = tid & 255;
    float v = (kk < HH) ? W_ih[j * HH + kk] : W_hh[j * HH + (kk - HH)];
    Wcat_h[tid] = (_Float16)v;
  }
  if (tid < 512) bias_g[tid] = b_ih[tid] + b_hh[tid];
}

__global__ void k_init(const float* __restrict__ h0, const float* __restrict__ c0,
                       float* __restrict__ hbuf, float* __restrict__ cbuf,
                       _Float16* __restrict__ xcat) {
  int tid = blockIdx.x * blockDim.x + threadIdx.x;
  if (tid < NN * HH) {
    float h = h0[tid];
    hbuf[tid] = h;
    cbuf[tid] = c0[tid];
    int n = tid >> 7, hh = tid & 127;
    xcat[n * 256 + 128 + hh] = (_Float16)h;
  }
}

// ---- per-step: input embedding, writes xcat[:, 0:64] ------------------------
__global__ void k_inp(const float* __restrict__ scene_t, const float* __restrict__ W_in,
                      const float* __restrict__ b_in, _Float16* __restrict__ xcat) {
  int tid = blockIdx.x * blockDim.x + threadIdx.x;
  if (tid >= NN * EE) return;
  int n = tid >> 6, e = tid & 63;
  float x0 = scene_t[n * 3 + 1];
  float x1 = scene_t[n * 3 + 2];
  float v = W_in[e * 2] * x0 + W_in[e * 2 + 1] * x1 + b_in[e];
  xcat[n * 256 + e] = (_Float16)(v > 0.0f ? v : 0.0f);
}

// ---- GEMM1: M[o,g,e] = h @ W_t_g^T, stored transposed: Mbuf[e][o*64+g] ------
// 1024 waves: (g, o-tile); 4 e-tile accumulators, pinned pipelined k-loop.
__global__ void __launch_bounds__(128) k_gemm1(const _Float16* __restrict__ xcat,
                                               const _Float16* __restrict__ Wt_h,
                                               _Float16* __restrict__ Mbuf) {
  int wave = blockIdx.x * 4 + (threadIdx.x >> 5);
  int lane = threadIdx.x & 31;
  int ot = wave & 15, g = wave >> 4;
  int mloc = lane & 15, khalf = lane >> 4;
  const _Float16* A   = xcat + 128;                              // h part, lda=256
  const _Float16* BT0 = Wt_h + (size_t)0  * (G2 * HH) + g * HH;  // ldb=8192
  const _Float16* BT1 = Wt_h + (size_t)16 * (G2 * HH) + g * HH;
  const _Float16* BT2 = Wt_h + (size_t)32 * (G2 * HH) + g * HH;
  const _Float16* BT3 = Wt_h + (size_t)48 * (G2 * HH) + g * HH;
  v8f acc0 = {}, acc1 = {}, acc2 = {}, acc3 = {};
  v16h a  = ld_a(A, 256, ot * 16 + mloc, khalf, 0);
  v16h b0 = ld_b(BT0, G2 * HH, mloc, khalf, 0);
  v16h b1 = ld_b(BT1, G2 * HH, mloc, khalf, 0);
  v16h b2 = ld_b(BT2, G2 * HH, mloc, khalf, 0);
  v16h b3 = ld_b(BT3, G2 * HH, mloc, khalf, 0);
#pragma unroll
  for (int k = 0; k < HH - 32; k += 32) {
    v16h an  = ld_a(A, 256, ot * 16 + mloc, khalf, k + 32);
    v16h bn0 = ld_b(BT0, G2 * HH, mloc, khalf, k + 32);
    v16h bn1 = ld_b(BT1, G2 * HH, mloc, khalf, k + 32);
    v16h bn2 = ld_b(BT2, G2 * HH, mloc, khalf, k + 32);
    v16h bn3 = ld_b(BT3, G2 * HH, mloc, khalf, k + 32);
    acc0 = WMMA_F16(a, b0, acc0);
    acc1 = WMMA_F16(a, b1, acc1);
    acc2 = WMMA_F16(a, b2, acc2);
    acc3 = WMMA_F16(a, b3, acc3);
    SCHED_GROUP(SG_VMEM_RD, 10);   // iter n+1 loads first ...
    SCHED_GROUP(SG_WMMA, 4);       // ... then iter n WMMAs
    a = an; b0 = bn0; b1 = bn1; b2 = bn2; b3 = bn3;
  }
  acc0 = WMMA_F16(a, b0, acc0);
  acc1 = WMMA_F16(a, b1, acc1);
  acc2 = WMMA_F16(a, b2, acc2);
  acc3 = WMMA_F16(a, b3, acc3);
  v8f accs[4] = {acc0, acc1, acc2, acc3};
#pragma unroll
  for (int et = 0; et < 4; ++et) {
    int e = et * 16 + mloc;
#pragma unroll
    for (int r = 0; r < 8; ++r) {
      int o = ot * 16 + r + khalf * 8;
      Mbuf[(size_t)e * KSOC + o * G2 + g] = (_Float16)accs[et][r];  // transposed store
    }
  }
}

// ---- GEMM2: ten_part[c] = grid_f32(256x16384) @ Mbuf^T, K split 64 ways -----
// Fused f32->f16 on the A stream (single HBM pass over grids[t]).
__global__ void __launch_bounds__(128) k_gemm2(const float* __restrict__ grid_t,
                                               const _Float16* __restrict__ Mbuf,
                                               float* __restrict__ ten_part) {
  int wave = blockIdx.x * 4 + (threadIdx.x >> 5);
  int lane = threadIdx.x & 31;
  int ot = wave & 15, ch = wave >> 4;                           // ch: 0..63
  int mloc = lane & 15, khalf = lane >> 4;
  int k0 = ch * KCHUNK;
  const float* Arow = grid_t + (size_t)(ot * 16 + mloc) * KSOC + k0 + khalf * 8;
  const _Float16* BT0 = Mbuf + (size_t)0  * KSOC;
  const _Float16* BT1 = Mbuf + (size_t)16 * KSOC;
  const _Float16* BT2 = Mbuf + (size_t)32 * KSOC;
  const _Float16* BT3 = Mbuf + (size_t)48 * KSOC;
  v8f acc0 = {}, acc1 = {}, acc2 = {}, acc3 = {};
  v16h a  = ld_a_f32(Arow);
  v16h b0 = ld_b(BT0, KSOC, mloc, khalf, k0);
  v16h b1 = ld_b(BT1, KSOC, mloc, khalf, k0);
  v16h b2 = ld_b(BT2, KSOC, mloc, khalf, k0);
  v16h b3 = ld_b(BT3, KSOC, mloc, khalf, k0);
#pragma unroll
  for (int k = 0; k < KCHUNK - 32; k += 32) {
    v16h an  = ld_a_f32(Arow + k + 32);
    v16h bn0 = ld_b(BT0, KSOC, mloc, khalf, k0 + k + 32);
    v16h bn1 = ld_b(BT1, KSOC, mloc, khalf, k0 + k + 32);
    v16h bn2 = ld_b(BT2, KSOC, mloc, khalf, k0 + k + 32);
    v16h bn3 = ld_b(BT3, KSOC, mloc, khalf, k0 + k + 32);
    acc0 = WMMA_F16(a, b0, acc0);
    acc1 = WMMA_F16(a, b1, acc1);
    acc2 = WMMA_F16(a, b2, acc2);
    acc3 = WMMA_F16(a, b3, acc3);
    SCHED_GROUP(SG_VMEM_RD, 12);   // iter n+1 loads (4xA f32 + 8xB f16) first
    SCHED_GROUP(SG_WMMA, 4);       // ... then iter n WMMAs
    a = an; b0 = bn0; b1 = bn1; b2 = bn2; b3 = bn3;
  }
  acc0 = WMMA_F16(a, b0, acc0);
  acc1 = WMMA_F16(a, b1, acc1);
  acc2 = WMMA_F16(a, b2, acc2);
  acc3 = WMMA_F16(a, b3, acc3);
  float* dst = ten_part + (size_t)ch * (NN * EE);
  v8f accs[4] = {acc0, acc1, acc2, acc3};
#pragma unroll
  for (int et = 0; et < 4; ++et) {
    int e = et * 16 + mloc;
#pragma unroll
    for (int r = 0; r < 8; ++r) {
      int p = ot * 16 + r + khalf * 8;
      dst[p * EE + e] = accs[et][r];                            // deterministic partials
    }
  }
}

// ---- reduce partials + bias + relu -> xcat[:, 64:128] -----------------------
__global__ void k_ten(const float* __restrict__ ten_part, const float* __restrict__ b_t,
                      _Float16* __restrict__ xcat) {
  int tid = blockIdx.x * blockDim.x + threadIdx.x;
  if (tid >= NN * EE) return;
  int n = tid >> 6, e = tid & 63;
  float s = b_t[e];
#pragma unroll 8
  for (int c = 0; c < NCHUNK; ++c) s += ten_part[c * (NN * EE) + tid];
  xcat[n * 256 + 64 + e] = (_Float16)(s > 0.0f ? s : 0.0f);
}

// ---- GEMM3: gates = xcat(256x256) @ Wcat^T(512x256) + bias ------------------
__global__ void __launch_bounds__(128) k_gemm3(const _Float16* __restrict__ xcat,
                                               const _Float16* __restrict__ Wcat_h,
                                               const float* __restrict__ bias_g,
                                               float* __restrict__ gates) {
  int wave = blockIdx.x * 4 + (threadIdx.x >> 5);
  int lane = threadIdx.x & 31;
  int jt = wave & 31, pt = wave >> 5;
  int mloc = lane & 15, khalf = lane >> 4;
  const _Float16* BT = Wcat_h + (size_t)(jt * 16) * 256;
  v8f acc = {};
  v16h a = ld_a(xcat, 256, pt * 16 + mloc, khalf, 0);
  v16h b = ld_b(BT, 256, mloc, khalf, 0);
#pragma unroll
  for (int k = 0; k < 256 - 32; k += 32) {
    v16h an = ld_a(xcat, 256, pt * 16 + mloc, khalf, k + 32);
    v16h bn = ld_b(BT, 256, mloc, khalf, k + 32);
    acc = WMMA_F16(a, b, acc);
    SCHED_GROUP(SG_VMEM_RD, 4);
    SCHED_GROUP(SG_WMMA, 1);
    a = an; b = bn;
  }
  acc = WMMA_F16(a, b, acc);
  int j = jt * 16 + mloc;
  float bias = bias_g[j];
#pragma unroll
  for (int r = 0; r < 8; ++r) {
    int p = pt * 16 + r + khalf * 8;
    gates[p * 512 + j] = acc[r] + bias;
  }
}

// ---- LSTM elementwise: c,h update; refresh xcat h-columns -------------------
__global__ void k_lstm(const float* __restrict__ gates, float* __restrict__ cbuf,
                       float* __restrict__ hbuf, _Float16* __restrict__ xcat) {
  int tid = blockIdx.x * blockDim.x + threadIdx.x;
  if (tid >= NN * HH) return;
  int n = tid >> 7, hh = tid & 127;
  const float* gr = gates + n * 512;
  float ig = gr[hh], fg = gr[128 + hh], gg = gr[256 + hh], og = gr[384 + hh];
  float c2 = sigmf(fg) * cbuf[tid] + sigmf(ig) * tanhf(gg);
  float h2 = sigmf(og) * tanhf(c2);
  cbuf[tid] = c2;
  hbuf[tid] = h2;
  xcat[n * 256 + 128 + hh] = (_Float16)h2;
}

// ---- output head: out_t = h2 @ W_out^T + b_out ------------------------------
__global__ void k_out(const float* __restrict__ hbuf, const float* __restrict__ W_out,
                      const float* __restrict__ b_out, float* __restrict__ out_t) {
  int tid = blockIdx.x * blockDim.x + threadIdx.x;
  if (tid >= NN * 5) return;
  int n = tid / 5, j = tid % 5;
  float s = b_out[j];
  for (int k = 0; k < HH; ++k) s += hbuf[n * HH + k] * W_out[j * HH + k];
  out_t[n * 5 + j] = s;
}

__global__ void k_fin(const float* __restrict__ hbuf, const float* __restrict__ cbuf,
                      float* __restrict__ out_tail) {
  int tid = blockIdx.x * blockDim.x + threadIdx.x;
  if (tid < NN * HH) out_tail[tid] = hbuf[tid];
  else if (tid < 2 * NN * HH) out_tail[tid] = cbuf[tid - NN * HH];
}

extern "C" void kernel_launch(void* const* d_in, const int* in_sizes, int n_in,
                              void* d_out, int out_size, void* d_ws, size_t ws_size,
                              hipStream_t stream) {
  const float* scene = (const float*)d_in[0];
  const float* grids = (const float*)d_in[1];
  const float* h0    = (const float*)d_in[2];
  const float* c0    = (const float*)d_in[3];
  // d_in[4] = dimensions (unused)
  const float* W_in  = (const float*)d_in[5];
  const float* b_in  = (const float*)d_in[6];
  const float* W_t   = (const float*)d_in[7];
  const float* b_t   = (const float*)d_in[8];
  const float* W_ih  = (const float*)d_in[9];
  const float* b_ih  = (const float*)d_in[10];
  const float* W_hh  = (const float*)d_in[11];
  const float* b_hh  = (const float*)d_in[12];
  const float* W_out = (const float*)d_in[13];
  const float* b_out = (const float*)d_in[14];
  float* out = (float*)d_out;

  char* ws = (char*)d_ws;
  size_t off = 0;
  auto carve = [&](size_t bytes) -> void* {
    void* p = ws + off;
    off = (off + bytes + 255) & ~(size_t)255;
    return p;
  };
  _Float16* Wt_h     = (_Float16*)carve((size_t)EE * G2 * HH * 2);   // 1 MB
  _Float16* Wcat_h   = (_Float16*)carve((size_t)512 * 256 * 2);      // 256 KB
  float*    bias_g   = (float*)   carve(512 * 4);
  float*    hbuf     = (float*)   carve((size_t)NN * HH * 4);
  float*    cbuf     = (float*)   carve((size_t)NN * HH * 4);
  _Float16* xcat     = (_Float16*)carve((size_t)NN * 256 * 2);
  _Float16* Mbuf     = (_Float16*)carve((size_t)EE * KSOC * 2);      // 2 MB (L2-resident)
  float*    ten_part = (float*)   carve((size_t)NCHUNK * NN * EE * 4); // 4 MB
  float*    gates    = (float*)   carve((size_t)NN * 512 * 4);

  k_prep<<<2048, 256, 0, stream>>>(W_t, W_ih, W_hh, b_ih, b_hh, Wt_h, Wcat_h, bias_g);
  k_init<<<128, 256, 0, stream>>>(h0, c0, hbuf, cbuf, xcat);

  for (int t = 0; t < TT; ++t) {
    const float* grid_t  = grids + (size_t)t * NN * KSOC;
    const float* scene_t = scene + (size_t)t * NN * 3;
    k_inp   <<<64,  256, 0, stream>>>(scene_t, W_in, b_in, xcat);
    k_gemm1 <<<256, 128, 0, stream>>>(xcat, Wt_h, Mbuf);
    k_gemm2 <<<256, 128, 0, stream>>>(grid_t, Mbuf, ten_part);
    k_ten   <<<64,  256, 0, stream>>>(ten_part, b_t, xcat);
    k_gemm3 <<<128, 128, 0, stream>>>(xcat, Wcat_h, bias_g, gates);
    k_lstm  <<<128, 256, 0, stream>>>(gates, cbuf, hbuf, xcat);
    k_out   <<<5,   256, 0, stream>>>(hbuf, W_out, b_out, out + (size_t)t * NN * 5);
  }
  k_fin<<<256, 256, 0, stream>>>(hbuf, cbuf, out + (size_t)TT * NN * 5);
}
